// HybridGeoNet_17343077941963
// MI455X (gfx1250) — matrix operand order repaired and checked by reading, not compiled, not run on hardware
//
#include <hip/hip_runtime.h>
#include <hip/hip_bf16.h>

// ---------------------------------------------------------------------------
// HybridGeoNet forward for MI455X (gfx1250, wave32, WMMA f16 16x16x32).
// All GEMMs run through v_wmma_f32_16x16x32_f16 with f32 accumulation.
// GEMM A-tiles staged via GLOBAL_LOAD_ASYNC_TO_LDS_B128 (ASYNCcnt path);
// global_prefetch_b8 for next tiles.
// ---------------------------------------------------------------------------

typedef _Float16 half_t;
typedef __attribute__((ext_vector_type(16))) _Float16 v16h;
typedef __attribute__((ext_vector_type(8)))  _Float16 h8;
typedef __attribute__((ext_vector_type(8)))  float    v8f;

// exact pointee type expected by __builtin_amdgcn_global_load_async_to_lds_b128
typedef int v4i_ __attribute__((vector_size(4 * sizeof(int))));
typedef __attribute__((address_space(1))) v4i_ g4i;   // global
typedef __attribute__((address_space(3))) v4i_ l4i;   // LDS

#define BB   4
#define CC   2048
#define SS   1024
#define EE   768
#define NN   4096      // BB*SS
#define NHD  12
#define DH   64
#define GHD  4
#define DG   192

#if defined(__has_builtin)
#  if __has_builtin(__builtin_amdgcn_global_load_async_to_lds_b128)
#    define HAVE_ASYNC_B128 1
#  else
#    define HAVE_ASYNC_B128 0
#  endif
#  if __has_builtin(__builtin_amdgcn_s_wait_asynccnt)
#    define HAVE_WAIT_ASYNC 1
#  else
#    define HAVE_WAIT_ASYNC 0
#  endif
#else
#  define HAVE_ASYNC_B128 0
#  define HAVE_WAIT_ASYNC 0
#endif

__device__ inline void wait_async0() {
#if HAVE_WAIT_ASYNC
  __builtin_amdgcn_s_wait_asynccnt(0);
#else
  asm volatile("s_wait_asynccnt 0" ::: "memory");
#endif
}

__device__ inline v8f zero8() {
  v8f z;
  #pragma unroll
  for (int i = 0; i < 8; i++) z[i] = 0.0f;
  return z;
}

// ---------------------------------------------------------------------------
// f32 -> f16 conversion (weights)
// ---------------------------------------------------------------------------
__global__ void f2h_kernel(const float* __restrict__ in, half_t* __restrict__ out, int n) {
  int i = blockIdx.x * 256 + threadIdx.x;
  if (i < n) out[i] = (half_t)in[i];
}

// ---------------------------------------------------------------------------
// LN over C=2048 for tokens gathered from x[B,C,H,W]; writes f16 [token][C]
// ---------------------------------------------------------------------------
__global__ __launch_bounds__(256) void ln_tok_kernel(
    const float* __restrict__ x, const float* __restrict__ g,
    const float* __restrict__ bt, half_t* __restrict__ out) {
  __shared__ float buf[CC];
  __shared__ float red[256];
  const int t = blockIdx.x;            // token index
  const int b = t >> 10, s = t & 1023;
  const int tid = threadIdx.x;
  float sum = 0.0f;
  for (int c = tid; c < CC; c += 256) {
    float v = x[((size_t)(b * CC + c) << 10) + s];
    buf[c] = v; sum += v;
  }
  red[tid] = sum; __syncthreads();
  for (int o = 128; o > 0; o >>= 1) { if (tid < o) red[tid] += red[tid + o]; __syncthreads(); }
  const float mean = red[0] * (1.0f / CC);
  __syncthreads();
  float sq = 0.0f;
  for (int c = tid; c < CC; c += 256) { float d = buf[c] - mean; sq += d * d; }
  red[tid] = sq; __syncthreads();
  for (int o = 128; o > 0; o >>= 1) { if (tid < o) red[tid] += red[tid + o]; __syncthreads(); }
  const float rstd = rsqrtf(red[0] * (1.0f / CC) + 1e-5f);
  for (int c = tid; c < CC; c += 256)
    out[(size_t)t * CC + c] = (half_t)((buf[c] - mean) * rstd * g[c] + bt[c]);
}

// ---------------------------------------------------------------------------
// LN over E=768 rows (f32 in); optional f16 and f32 outputs
// ---------------------------------------------------------------------------
__global__ __launch_bounds__(256) void ln_row_kernel(
    const float* __restrict__ in, const float* __restrict__ g,
    const float* __restrict__ bt, half_t* __restrict__ outh, float* __restrict__ outf) {
  __shared__ float red[256];
  const int t = blockIdx.x, tid = threadIdx.x;
  const float* row = in + (size_t)t * EE;
  float v0 = row[tid], v1 = row[tid + 256], v2 = row[tid + 512];
  red[tid] = v0 + v1 + v2; __syncthreads();
  for (int o = 128; o > 0; o >>= 1) { if (tid < o) red[tid] += red[tid + o]; __syncthreads(); }
  const float mean = red[0] * (1.0f / EE);
  __syncthreads();
  float d0 = v0 - mean, d1 = v1 - mean, d2 = v2 - mean;
  red[tid] = d0 * d0 + d1 * d1 + d2 * d2; __syncthreads();
  for (int o = 128; o > 0; o >>= 1) { if (tid < o) red[tid] += red[tid + o]; __syncthreads(); }
  const float rstd = rsqrtf(red[0] * (1.0f / EE) + 1e-5f);
  float y0 = d0 * rstd * g[tid]       + bt[tid];
  float y1 = d1 * rstd * g[tid + 256] + bt[tid + 256];
  float y2 = d2 * rstd * g[tid + 512] + bt[tid + 512];
  size_t o0 = (size_t)t * EE + tid;
  if (outh) { outh[o0] = (half_t)y0; outh[o0 + 256] = (half_t)y1; outh[o0 + 512] = (half_t)y2; }
  if (outf) { outf[o0] = y0; outf[o0 + 256] = y1; outf[o0 + 512] = y2; }
}

// ---------------------------------------------------------------------------
// Tiled WMMA GEMM: C[M,N] = act(A[M,K] * B[K,N] + bias (+ res))
// BM=128 BN=64 BK=32; 8 waves (2x4), each wave owns 32x32 (2x2 16x16 frags).
// A f16 row-major, B f16 row-major (transposed into LDS for frag loads).
// A tile staged via async global->LDS (ASYNCcnt) when available.
// ---------------------------------------------------------------------------
template <bool GELU_ACT>
__global__ __launch_bounds__(256) void gemm_wmma(
    const half_t* __restrict__ A, const half_t* __restrict__ Bw,
    const float* __restrict__ bias, const float* __restrict__ res,
    float* __restrict__ Cf, half_t* __restrict__ Ch, int M, int N, int K) {
  __shared__ half_t As[128][48];   // row-major A tile, 96B rows (16B aligned frags)
  __shared__ half_t Bs[64][48];    // B tile transposed: Bs[n][k]
  const int tid = threadIdx.x;
  const int lane = tid & 31, wave = tid >> 5;
  const int hi = lane >> 4, lm = lane & 15;
  const int wm = wave & 3, wn = wave >> 2;
  const int bm0 = blockIdx.y * 128, bn0 = blockIdx.x * 64;

  v8f acc[2][2];
  #pragma unroll
  for (int i = 0; i < 2; i++)
    #pragma unroll
    for (int j = 0; j < 2; j++) acc[i][j] = zero8();

  const int arow = tid >> 1, aseg = tid & 1;  // A staging: 128 rows x 2 segs of 16h
  const int bk = tid >> 3, bns = tid & 7;     // B staging: 32 k-rows x 8 segs of 8h

  for (int k0 = 0; k0 < K; k0 += 32) {
    __syncthreads();
    { // stage A (128x32): two 16-byte chunks per thread
      const half_t* src = A + (size_t)(bm0 + arow) * K + k0 + aseg * 16;
#if HAVE_ASYNC_B128
      __builtin_amdgcn_global_load_async_to_lds_b128(
          (g4i*)src, (l4i*)&As[arow][aseg * 16], 0, 0);
      __builtin_amdgcn_global_load_async_to_lds_b128(
          (g4i*)(src + 8), (l4i*)&As[arow][aseg * 16 + 8], 0, 0);
#else
      h8 v0 = *(const h8*)src;
      h8 v1 = *(const h8*)(src + 8);
      *(h8*)&As[arow][aseg * 16]     = v0;
      *(h8*)&As[arow][aseg * 16 + 8] = v1;
#endif
    }
    { // stage B transposed (32x64 -> Bs[n][k])
      const half_t* src = Bw + (size_t)(k0 + bk) * N + bn0 + bns * 8;
      h8 v = *(const h8*)src;
      #pragma unroll
      for (int j = 0; j < 8; j++) Bs[bns * 8 + j][bk] = v[j];
    }
    // prefetch next k-step tiles into L2/WGP$ (global_prefetch_b8)
    if (k0 + 32 < K) {
      __builtin_prefetch(A + (size_t)(bm0 + arow) * K + k0 + 32 + aseg * 16, 0, 1);
      __builtin_prefetch(Bw + (size_t)(k0 + 32 + bk) * N + bn0 + bns * 8, 0, 1);
    }
#if HAVE_ASYNC_B128
    wait_async0();
#endif
    __syncthreads();

    v16h af[2], bf[2];
    #pragma unroll
    for (int am = 0; am < 2; am++) {   // A frag: lane lm = row M; k = (j&7)+((j>>3)<<4)+(hi<<3)
      int r = wm * 32 + am * 16 + lm;
      h8 lo = *(const h8*)&As[r][hi * 8];
      h8 hh = *(const h8*)&As[r][16 + hi * 8];
      #pragma unroll
      for (int j = 0; j < 8; j++) { af[am][j] = lo[j]; af[am][8 + j] = hh[j]; }
    }
    #pragma unroll
    for (int an = 0; an < 2; an++) {   // B frag: lane lm = col N; k = (j&7)+((j>>3)<<4)+(hi<<3)
      int n = wn * 32 + an * 16 + lm;
      h8 lo = *(const h8*)&Bs[n][hi * 8];
      h8 hh = *(const h8*)&Bs[n][16 + hi * 8];
      #pragma unroll
      for (int j = 0; j < 8; j++) { bf[an][j] = lo[j]; bf[an][8 + j] = hh[j]; }
    }
    #pragma unroll
    for (int am = 0; am < 2; am++)
      #pragma unroll
      for (int an = 0; an < 2; an++)
        acc[am][an] = __builtin_amdgcn_wmma_f32_16x16x32_f16(
            false, af[am], false, bf[an], (short)0, acc[am][an], false, false);
  }

  // epilogue: C frag element (v, lane): row = v + 8*hi, col = lm
  #pragma unroll
  for (int am = 0; am < 2; am++) {
    #pragma unroll
    for (int an = 0; an < 2; an++) {
      const int gcol = bn0 + wn * 32 + an * 16 + lm;
      const float bv = bias ? bias[gcol] : 0.0f;
      #pragma unroll
      for (int v = 0; v < 8; v++) {
        const int grow = bm0 + wm * 32 + am * 16 + v + hi * 8;
        const size_t o = (size_t)grow * N + gcol;
        float xv = acc[am][an][v] + bv;
        if (res) xv += res[o];
        if (GELU_ACT) xv = 0.5f * xv * (1.0f + erff(xv * 0.70710678118654752f));
        if (Cf) Cf[o] = xv;
        if (Ch) Ch[o] = (half_t)xv;
      }
    }
  }
}

// ---------------------------------------------------------------------------
// Repack qkv f32 [token][3E] -> q/k/v f16 [B,H,S,dh]; q pre-scaled by 1/sqrt(64)
// ---------------------------------------------------------------------------
__global__ void repack_qkv_kernel(const float* __restrict__ qkv,
                                  half_t* __restrict__ q, half_t* __restrict__ k,
                                  half_t* __restrict__ v) {
  int i = blockIdx.x * 256 + threadIdx.x;
  if (i >= NN * EE) return;
  int t = i / EE, j = i - t * EE;
  int b = t >> 10, s = t & 1023;
  int h = j / DH, d = j - h * DH;
  size_t src = (size_t)t * (3 * EE) + j;
  size_t dst = ((size_t)(b * NHD + h) * SS + s) * DH + d;
  q[dst] = (half_t)(qkv[src] * 0.125f);
  k[dst] = (half_t)(qkv[src + EE]);
  v[dst] = (half_t)(qkv[src + 2 * EE]);
}

// ---------------------------------------------------------------------------
// Flash attention: one block per (128 q-rows, head, batch); 8 waves x 16 rows.
// S=QK^T via WMMA, online softmax (half-wave shuffles), P->LDS->A-frag, O=PV.
// ---------------------------------------------------------------------------
__global__ __launch_bounds__(256) void attn_flash_kernel(
    const half_t* __restrict__ Q, const half_t* __restrict__ Kt,
    const half_t* __restrict__ V, half_t* __restrict__ O) {
  __shared__ half_t Pl[8][16][64];     // per-wave P tile (16 q x 64 keys)
  const int tid = threadIdx.x;
  const int lane = tid & 31, wave = tid >> 5;
  const int hi = lane >> 4, lm = lane & 15;
  const int h = blockIdx.y, b = blockIdx.z;
  const int bh = b * NHD + h;
  const int q0 = blockIdx.x * 128 + wave * 16;
  const half_t* Qb = Q + (size_t)bh * SS * DH;
  const half_t* Kb = Kt + (size_t)bh * SS * DH;
  const half_t* Vb = V + (size_t)bh * SS * DH;

  v16h qf[2];
  #pragma unroll
  for (int ks = 0; ks < 2; ks++) {
    const half_t* src = Qb + (size_t)(q0 + lm) * DH + ks * 32;
    h8 lo = *(const h8*)(src + hi * 8);
    h8 hh = *(const h8*)(src + 16 + hi * 8);
    #pragma unroll
    for (int j = 0; j < 8; j++) { qf[ks][j] = lo[j]; qf[ks][8 + j] = hh[j]; }
  }

  float mrow[8], lrow[8];
  v8f oacc[4];
  #pragma unroll
  for (int v = 0; v < 8; v++) { mrow[v] = -3.0e38f; lrow[v] = 0.0f; }
  #pragma unroll
  for (int nt = 0; nt < 4; nt++) oacc[nt] = zero8();

  for (int kt = 0; kt < 16; kt++) {
    const int key0 = kt * 64;
    v8f sacc[4];
    #pragma unroll
    for (int nt = 0; nt < 4; nt++) sacc[nt] = zero8();
    #pragma unroll
    for (int nt = 0; nt < 4; nt++) {
      const int key = key0 + nt * 16 + lm;
      #pragma unroll
      for (int ks = 0; ks < 2; ks++) {
        v16h bf;
        const half_t* src = Kb + (size_t)key * DH + ks * 32 + hi * 8;
        h8 lo = *(const h8*)src;
        h8 hh = *(const h8*)(src + 16);
        #pragma unroll
        for (int j = 0; j < 8; j++) { bf[j] = lo[j]; bf[8 + j] = hh[j]; }
        sacc[nt] = __builtin_amdgcn_wmma_f32_16x16x32_f16(
            false, qf[ks], false, bf, (short)0, sacc[nt], false, false);
      }
    }
    // online softmax: row (v + 8*hi) lives across the 16 lanes of this half
    #pragma unroll
    for (int v = 0; v < 8; v++) {
      float mx = fmaxf(fmaxf(sacc[0][v], sacc[1][v]), fmaxf(sacc[2][v], sacc[3][v]));
      for (int off = 1; off < 16; off <<= 1) mx = fmaxf(mx, __shfl_xor(mx, off, 16));
      const float mnew = fmaxf(mrow[v], mx);
      const float corr = __expf(mrow[v] - mnew);
      mrow[v] = mnew;
      lrow[v] *= corr;
      #pragma unroll
      for (int nt = 0; nt < 4; nt++) oacc[nt][v] *= corr;
      float rs = 0.0f;
      #pragma unroll
      for (int nt = 0; nt < 4; nt++) {
        float p = __expf(sacc[nt][v] - mnew);
        sacc[nt][v] = p; rs += p;
      }
      for (int off = 1; off < 16; off <<= 1) rs += __shfl_xor(rs, off, 16);
      lrow[v] += rs;
    }
    // P -> LDS (C-layout to row-major), same-wave LDS ops stay in order
    #pragma unroll
    for (int nt = 0; nt < 4; nt++)
      #pragma unroll
      for (int v = 0; v < 8; v++)
        Pl[wave][v + hi * 8][nt * 16 + lm] = (half_t)sacc[nt][v];
    // O += P * V
    #pragma unroll
    for (int ks2 = 0; ks2 < 2; ks2++) {
      v16h pf;
      {
        const half_t* src = &Pl[wave][lm][ks2 * 32];
        h8 lo = *(const h8*)(src + hi * 8);
        h8 hh = *(const h8*)(src + 16 + hi * 8);
        #pragma unroll
        for (int j = 0; j < 8; j++) { pf[j] = lo[j]; pf[8 + j] = hh[j]; }
      }
      #pragma unroll
      for (int nt = 0; nt < 4; nt++) {
        v16h vf;
        #pragma unroll
        for (int j = 0; j < 16; j++) {
          const int key = key0 + ks2 * 32 + (j & 7) + ((j >> 3) << 4) + hi * 8;
          vf[j] = Vb[(size_t)key * DH + nt * 16 + lm];
        }
        oacc[nt] = __builtin_amdgcn_wmma_f32_16x16x32_f16(
            false, pf, false, vf, (short)0, oacc[nt], false, false);
      }
    }
  }
  // normalize and write o16 [token][E] with head-interleaved columns
  #pragma unroll
  for (int nt = 0; nt < 4; nt++)
    #pragma unroll
    for (int v = 0; v < 8; v++) {
      const int row = q0 + v + hi * 8;
      O[(size_t)(b * SS + row) * EE + h * DH + nt * 16 + lm] =
          (half_t)(oacc[nt][v] / lrow[v]);
    }
}

// ---------------------------------------------------------------------------
// GAT attention logits a_s/a_d: one wave per (node, head)
// ---------------------------------------------------------------------------
__global__ __launch_bounds__(256) void asad_kernel(
    const float* __restrict__ hN, const float* __restrict__ asrc,
    const float* __restrict__ adst, float* __restrict__ AS, float* __restrict__ AD) {
  const int gid = blockIdx.x * 256 + threadIdx.x;
  const int w = gid >> 5, lane = gid & 31;
  if (w >= NN * GHD) return;
  const int n = w >> 2, h = w & 3;
  float s = 0.0f, d = 0.0f;
  for (int dd = lane; dd < DG; dd += 32) {
    const float v = hN[(size_t)n * EE + h * DG + dd];
    s += v * asrc[h * DG + dd];
    d += v * adst[h * DG + dd];
  }
  for (int off = 16; off > 0; off >>= 1) { s += __shfl_xor(s, off, 32); d += __shfl_xor(d, off, 32); }
  if (lane == 0) { AS[n * GHD + h] = s; AD[n * GHD + h] = d; }
}

// ---------------------------------------------------------------------------
// GAT aggregation: static grid 8-neighborhood + self-loop, direct gather
// (matches the reference edge builder incl. asymmetric down-left diagonal).
// ---------------------------------------------------------------------------
__global__ __launch_bounds__(256) void gat_agg_kernel(
    const float* __restrict__ hN, const float* __restrict__ AS,
    const float* __restrict__ AD, const float* __restrict__ xattn,
    const float* __restrict__ gb, float* __restrict__ xafter) {
  __shared__ int srcs[9];
  __shared__ int scnt;
  __shared__ float al[GHD][9];
  const int i = blockIdx.x;
  const int tid = threadIdx.x;
  if (tid == 0) {
    const int b = i >> 10, s = i & 1023, r = s >> 5, c = s & 31;
    const int base = b * SS;
    int cnt = 0;
    auto add = [&](int rr, int cc) { srcs[cnt++] = base + rr * 32 + cc; };
    if (c >= 1)              add(r, c - 1);        // horizontal fwd incoming
    if (c <= 30)             add(r, c + 1);        // horizontal rev
    if (r >= 1)              add(r - 1, c);        // vertical fwd
    if (r <= 30)             add(r + 1, c);        // vertical rev
    if (r >= 1 && c >= 1)    add(r - 1, c - 1);    // down-right fwd
    if (r <= 30 && c <= 30)  add(r + 1, c + 1);    // down-right rev
    if (r >= 1 && c <= 29)   add(r - 1, c + 1);    // down-left fwd (c+1 in [1,30])
    if (r <= 30 && c >= 1 && c <= 30) add(r + 1, c - 1); // down-left rev
    add(r, c);                                     // self loop
    scnt = cnt;
  }
  __syncthreads();
  const int cnt = scnt;
  if (tid < GHD * 9) {
    const int h = tid / 9, j = tid - h * 9;
    if (j < cnt) {
      float e = AS[srcs[j] * GHD + h] + AD[i * GHD + h];
      al[h][j] = (e > 0.0f) ? e : 0.2f * e;        // leaky_relu(0.2)
    }
  }
  __syncthreads();
  if (tid < GHD) {
    const int h = tid;
    float mx = -3.0e38f;
    for (int j = 0; j < cnt; j++) mx = fmaxf(mx, al[h][j]);
    float den = 0.0f;
    for (int j = 0; j < cnt; j++) { float e = __expf(al[h][j] - mx); al[h][j] = e; den += e; }
    const float inv = 1.0f / den;
    for (int j = 0; j < cnt; j++) al[h][j] *= inv;
  }
  __syncthreads();
  for (int ch = tid; ch < EE; ch += 256) {
    const int h = ch / DG;
    float acc = 0.0f;
    for (int j = 0; j < cnt; j++) acc += al[h][j] * hN[(size_t)srcs[j] * EE + ch];
    acc += gb[ch];
    const float gat = (acc > 0.0f) ? acc : (__expf(acc) - 1.0f);   // elu
    xafter[(size_t)i * EE + ch] = xattn[(size_t)i * EE + ch] + gat;
  }
}

// ---------------------------------------------------------------------------
// Final: z = n3 + mlp; LN(z, fn); scatter to out[B,E,H,W]
// ---------------------------------------------------------------------------
__global__ __launch_bounds__(256) void final_ln_kernel(
    const float* __restrict__ n3, const float* __restrict__ mlp,
    const float* __restrict__ g, const float* __restrict__ bt,
    float* __restrict__ out) {
  __shared__ float red[256];
  const int t = blockIdx.x, tid = threadIdx.x;
  const int b = t >> 10, s = t & 1023;
  const size_t r0 = (size_t)t * EE;
  float v0 = n3[r0 + tid] + mlp[r0 + tid];
  float v1 = n3[r0 + tid + 256] + mlp[r0 + tid + 256];
  float v2 = n3[r0 + tid + 512] + mlp[r0 + tid + 512];
  red[tid] = v0 + v1 + v2; __syncthreads();
  for (int o = 128; o > 0; o >>= 1) { if (tid < o) red[tid] += red[tid + o]; __syncthreads(); }
  const float mean = red[0] * (1.0f / EE);
  __syncthreads();
  float d0 = v0 - mean, d1 = v1 - mean, d2 = v2 - mean;
  red[tid] = d0 * d0 + d1 * d1 + d2 * d2; __syncthreads();
  for (int o = 128; o > 0; o >>= 1) { if (tid < o) red[tid] += red[tid + o]; __syncthreads(); }
  const float rstd = rsqrtf(red[0] * (1.0f / EE) + 1e-5f);
  out[((size_t)(b * EE + tid)       << 10) + s] = d0 * rstd * g[tid]       + bt[tid];
  out[((size_t)(b * EE + tid + 256) << 10) + s] = d1 * rstd * g[tid + 256] + bt[tid + 256];
  out[((size_t)(b * EE + tid + 512) << 10) + s] = d2 * rstd * g[tid + 512] + bt[tid + 512];
}

// ---------------------------------------------------------------------------
extern "C" void kernel_launch(void* const* d_in, const int* in_sizes, int n_in,
                              void* d_out, int out_size, void* d_ws, size_t ws_size,
                              hipStream_t stream) {
  (void)in_sizes; (void)n_in; (void)out_size; (void)ws_size;
  const float* x      = (const float*)d_in[0];
  const float* ln1_g  = (const float*)d_in[1];
  const float* ln1_b  = (const float*)d_in[2];
  const float* proj_w = (const float*)d_in[3];
  const float* proj_b = (const float*)d_in[4];
  const float* wqkv   = (const float*)d_in[5];
  const float* bqkv   = (const float*)d_in[6];
  const float* wo     = (const float*)d_in[7];
  const float* bo     = (const float*)d_in[8];
  const float* ln2_g  = (const float*)d_in[9];
  const float* ln2_b  = (const float*)d_in[10];
  const float* gp_w   = (const float*)d_in[11];
  const float* gp_b   = (const float*)d_in[12];
  const float* gat_w  = (const float*)d_in[13];
  const float* attsrc = (const float*)d_in[14];
  const float* attdst = (const float*)d_in[15];
  const float* gat_b  = (const float*)d_in[16];
  const float* ln3_g  = (const float*)d_in[17];
  const float* ln3_b  = (const float*)d_in[18];
  const float* mlp_w1 = (const float*)d_in[19];
  const float* mlp_b1 = (const float*)d_in[20];
  const float* mlp_w2 = (const float*)d_in[21];
  const float* mlp_b2 = (const float*)d_in[22];
  const float* fn_g   = (const float*)d_in[23];
  const float* fn_b   = (const float*)d_in[24];
  float* out = (float*)d_out;

  char* ws = (char*)d_ws;
  size_t off = 0;
  auto alloc = [&](size_t bytes) -> char* {
    char* p = ws + off;
    off += (bytes + 255) & ~(size_t)255;
    return p;
  };

  // f16 weights
  half_t* projW16 = (half_t*)alloc((size_t)CC * EE * 2);
  half_t* wqkv16  = (half_t*)alloc((size_t)EE * 3 * EE * 2);
  half_t* wo16    = (half_t*)alloc((size_t)EE * EE * 2);
  half_t* gp16    = (half_t*)alloc((size_t)EE * EE * 2);
  half_t* gat16   = (half_t*)alloc((size_t)EE * EE * 2);
  half_t* w116    = (half_t*)alloc((size_t)EE * EE * 2);
  half_t* w216    = (half_t*)alloc((size_t)EE * EE * 2);
  // tokLN region (reused: o16 + ln2_16 after GEMM1 consumes it)
  half_t* tokLN16 = (half_t*)alloc((size_t)NN * CC * 2);
  half_t* o16     = tokLN16;
  half_t* ln2_16  = tokLN16 + (size_t)NN * EE;
  // activations
  float*  vinF    = (float*)alloc((size_t)NN * EE * 4);
  half_t* vin16   = (half_t*)alloc((size_t)NN * EE * 2);
  float*  qkvF    = (float*)alloc((size_t)NN * 3 * EE * 4);  // reused: hN + n3 + mlp
  float*  hNF     = qkvF;
  float*  n3F     = qkvF + (size_t)NN * EE;
  float*  mlpF    = qkvF + (size_t)2 * NN * EE;
  half_t* q16     = (half_t*)alloc((size_t)NN * EE * 2);
  half_t* k16     = (half_t*)alloc((size_t)NN * EE * 2);
  half_t* v16     = (half_t*)alloc((size_t)NN * EE * 2);
  float*  xattnF  = (float*)alloc((size_t)NN * EE * 4);
  half_t* g16     = (half_t*)alloc((size_t)NN * EE * 2);
  half_t* n316    = (half_t*)alloc((size_t)NN * EE * 2);
  half_t* h116    = (half_t*)alloc((size_t)NN * EE * 2);
  float*  xafterF = (float*)alloc((size_t)NN * EE * 4);
  float*  AS      = (float*)alloc((size_t)NN * GHD * 4);
  float*  AD      = (float*)alloc((size_t)NN * GHD * 4);

  auto cvt = [&](const float* src, half_t* dst, int n) {
    f2h_kernel<<<(n + 255) / 256, 256, 0, stream>>>(src, dst, n);
  };
  cvt(proj_w, projW16, CC * EE);
  cvt(wqkv,   wqkv16,  EE * 3 * EE);
  cvt(wo,     wo16,    EE * EE);
  cvt(gp_w,   gp16,    EE * EE);
  cvt(gat_w,  gat16,   EE * EE);
  cvt(mlp_w1, w116,    EE * EE);
  cvt(mlp_w2, w216,    EE * EE);

  // 1) LN1 over C, f16 tokens
  ln_tok_kernel<<<NN, 256, 0, stream>>>(x, ln1_g, ln1_b, tokLN16);
  // 2) v_in = tokLN @ proj_w + proj_b  (f32 + f16)
  gemm_wmma<false><<<dim3(EE / 64, NN / 128), 256, 0, stream>>>(
      tokLN16, projW16, proj_b, nullptr, vinF, vin16, NN, EE, CC);
  // 3) qkv = v_in @ wqkv + bqkv
  gemm_wmma<false><<<dim3(3 * EE / 64, NN / 128), 256, 0, stream>>>(
      vin16, wqkv16, bqkv, nullptr, qkvF, nullptr, NN, 3 * EE, EE);
  // 4) repack q/k/v into [B,H,S,dh] f16 (q scaled by 1/sqrt(dh))
  repack_qkv_kernel<<<(NN * EE + 255) / 256, 256, 0, stream>>>(qkvF, q16, k16, v16);
  // 5) fused flash attention
  attn_flash_kernel<<<dim3(SS / 128, NHD, BB), 256, 0, stream>>>(q16, k16, v16, o16);
  // 6) x_attn = v_in + o @ wo + bo
  gemm_wmma<false><<<dim3(EE / 64, NN / 128), 256, 0, stream>>>(
      o16, wo16, bo, vinF, xattnF, nullptr, NN, EE, EE);
  // 7) GAT: LN2 -> gp -> gat_w -> per-node grid gather softmax
  ln_row_kernel<<<NN, 256, 0, stream>>>(xattnF, ln2_g, ln2_b, ln2_16, nullptr);
  gemm_wmma<false><<<dim3(EE / 64, NN / 128), 256, 0, stream>>>(
      ln2_16, gp16, gp_b, nullptr, nullptr, g16, NN, EE, EE);
  gemm_wmma<false><<<dim3(EE / 64, NN / 128), 256, 0, stream>>>(
      g16, gat16, nullptr, nullptr, hNF, nullptr, NN, EE, EE);
  asad_kernel<<<(NN * GHD * 32 + 255) / 256, 256, 0, stream>>>(hNF, attsrc, attdst, AS, AD);
  gat_agg_kernel<<<NN, 256, 0, stream>>>(hNF, AS, AD, xattnF, gat_b, xafterF);
  // 8) MLP: LN3 -> gelu(w1) -> w2
  ln_row_kernel<<<NN, 256, 0, stream>>>(xafterF, ln3_g, ln3_b, n316, n3F);
  gemm_wmma<true><<<dim3(EE / 64, NN / 128), 256, 0, stream>>>(
      n316, w116, mlp_b1, nullptr, nullptr, h116, NN, EE, EE);
  gemm_wmma<false><<<dim3(EE / 64, NN / 128), 256, 0, stream>>>(
      h116, w216, mlp_b2, nullptr, mlpF, nullptr, NN, EE, EE);
  // 9) final LN + transpose to [B,E,H,W]
  final_ln_kernel<<<NN, 256, 0, stream>>>(n3F, mlpF, fn_g, fn_b, out);
}
